// MeshRefinementStage_7078106104228
// MI455X (gfx1250) — matrix operand order repaired and checked by reading, not compiled, not run on hardware
//
#include <hip/hip_runtime.h>

typedef __attribute__((ext_vector_type(16))) __bf16 v16bf;
typedef __attribute__((ext_vector_type(8)))  float  v8f;

#define HIDN   256
#define INK    259
#define KPAD   288   // 259 padded to multiple of 32; 288*2B stride is 16B-aligned
#define CCH    256
#define IMH    64
#define IMW    64

__device__ __forceinline__ unsigned short f2bf(float f) {
    unsigned int u = __float_as_uint(f);
    unsigned int r = u + 0x7FFFu + ((u >> 16) & 1u);   // round-to-nearest-even
    return (unsigned short)(r >> 16);
}

// ---------------- weight prep: f32 -> bf16, concat gw0/gw1 per layer, pad K to 288 -------------
__global__ void prep_weights(const float* __restrict__ bott_w,
                             const float* __restrict__ gw0, const float* __restrict__ gw1,
                             const float* __restrict__ gb0, const float* __restrict__ gb1,
                             unsigned short* __restrict__ bottbf,
                             unsigned short* __restrict__ layw,
                             float* __restrict__ layb) {
    int idx = blockIdx.x * blockDim.x + threadIdx.x;
    if (idx < HIDN * HIDN) bottbf[idx] = f2bf(bott_w[idx]);          // (256,256) row-major
    if (idx < 3 * 512 * KPAD) {                                       // (512,288) per layer
        int l = idx / (512 * KPAD);
        int rem = idx - l * 512 * KPAD;
        int r = rem / KPAD, k = rem - r * KPAD;
        float v = 0.f;
        if (k < INK)
            v = (r < HIDN) ? gw0[(l * HIDN + r) * INK + k]
                           : gw1[(l * HIDN + (r - HIDN)) * INK + k];
        layw[idx] = f2bf(v);
    }
    if (idx < 3 * 512) {
        int l = idx / 512, r = idx - l * 512;
        layb[idx] = (r < HIDN) ? gb0[l * HIDN + r] : gb1[l * HIDN + (r - HIDN)];
    }
}

// ---------------- vfeats (bf16) position columns + zero padding (written once) -----------------
__global__ void vfeats_init(const float* __restrict__ verts, unsigned short* __restrict__ vfeats,
                            int V) {
    int v = blockIdx.x * blockDim.x + threadIdx.x;
    if (v >= V) return;
    unsigned short* row = vfeats + (size_t)v * KPAD;
    row[256] = f2bf(verts[v * 3 + 0]);
    row[257] = f2bf(verts[v * 3 + 1]);
    row[258] = f2bf(verts[v * 3 + 2]);
#pragma unroll
    for (int k = INK; k < KPAD; ++k) row[k] = 0;
}

// ---------------- bilinear vert_align -> bf16 sampled features ---------------------------------
__global__ void sample_kernel(const float* __restrict__ x, const float* __restrict__ verts,
                              unsigned short* __restrict__ sampled, int V) {
    int v = blockIdx.x;
    int c = threadIdx.x;
    if (v >= V) return;
    float gx = verts[v * 3 + 0], gy = verts[v * 3 + 1];
    float ix = fminf(fmaxf((gx + 1.f) * 0.5f * (IMW - 1), 0.f), (float)(IMW - 1));
    float iy = fminf(fmaxf((gy + 1.f) * 0.5f * (IMH - 1), 0.f), (float)(IMH - 1));
    float x0 = floorf(ix), y0 = floorf(iy);
    float wx = ix - x0, wy = iy - y0;
    int x0i = (int)x0, y0i = (int)y0;
    int x1i = min(x0i + 1, IMW - 1), y1i = min(y0i + 1, IMH - 1);
    const float* xc = x + (size_t)c * (IMH * IMW);
    float v00 = xc[y0i * IMW + x0i];
    float v01 = xc[y0i * IMW + x1i];
    float v10 = xc[y1i * IMW + x0i];
    float v11 = xc[y1i * IMW + x1i];
    float val = (v00 * (1.f - wx) + v01 * wx) * (1.f - wy) +
                (v10 * (1.f - wx) + v11 * wx) * wy;
    sampled[(size_t)v * CCH + c] = f2bf(val);
}

// ---------------- bf16 WMMA GEMM: Y = act(Xbf @ Wb^T + bias) -----------------------------------
// block = 256 threads = 8 waves; block tile = 128(M) x 64(N).
// The block's 64xK weight slab (contiguous rows in Wb) is staged once into LDS and shared by all
// 8 waves (8x less L2 traffic for B).  Each wave owns a 16(M) x 64(N) strip: one A fragment
// (2 x global b128) per K step feeds 4 WMMAs (B via ds_load_b128) into 4 independent accumulators.
// MODE 0: single output, ReLU, bf16 write to Yb with stride KPAD (N=256)
// MODE 1: dual output (N=512): col<256 -> Y0 (f32, stride 256), col>=256 -> Y1 (f32, stride 256)
template <int LDX, int LDW, int MODE>
__global__ __launch_bounds__(256)
void gemm_wmma(const unsigned short* __restrict__ X,
               const unsigned short* __restrict__ Wb,
               const float* __restrict__ bias,
               unsigned short* __restrict__ Yb,
               float* __restrict__ Y0, float* __restrict__ Y1,
               int V) {
    __shared__ unsigned short ldsB[64 * LDW];      // 32KB (LDW=256) / 36KB (LDW=288)

    const int lane = threadIdx.x & 31;
    const int wave = threadIdx.x >> 5;
    const int row0 = blockIdx.x * 128 + wave * 16;
    const int col0 = blockIdx.y * 64;
    const int h = lane >> 4;        // lane half
    const int m = lane & 15;        // A row within tile / B column within tile

    // ---- cooperative fill: rows col0..col0+63 of Wb are contiguous -> straight block copy
    {
        const uint4* src = (const uint4*)(Wb + (size_t)col0 * LDW);
        uint4* dst = (uint4*)ldsB;
        const int NB = 64 * LDW / 8;               // uint4 count
#pragma unroll
        for (int i = threadIdx.x; i < NB; i += 256) dst[i] = src[i];
    }
    __syncthreads();

    int rowA = row0 + m;
    if (rowA >= V) rowA = V - 1;    // clamp loads; stores guarded below
    const unsigned short* arow = X + (size_t)rowA * LDX;

    const unsigned short* lb0 = ldsB + (0 * 16 + m) * LDW;
    const unsigned short* lb1 = ldsB + (1 * 16 + m) * LDW;
    const unsigned short* lb2 = ldsB + (2 * 16 + m) * LDW;
    const unsigned short* lb3 = ldsB + (3 * 16 + m) * LDW;

    v8f acc0 = {0.f, 0.f, 0.f, 0.f, 0.f, 0.f, 0.f, 0.f};
    v8f acc1 = acc0, acc2 = acc0, acc3 = acc0;

#pragma unroll
    for (int kb = 0; kb < LDW; kb += 32) {
        // ---- A fragment (global): us[0..15] <- K = h*16+0..15 (contiguous)
        union { uint4 q[2]; v16bf v; } A;
        A.q[0] = *(const uint4*)(arow + kb + h * 16);
        A.q[1] = *(const uint4*)(arow + kb + h * 16 + 8);
        // ---- B fragments (LDS): VGPR j <- K = h*16 + 2j, 2j+1 (contiguous 16 bf16 per lane)
        union { uint4 q[2]; v16bf v; } B0, B1, B2, B3;
        B0.q[0] = ((const uint4*)(lb0 + kb + h * 16))[0];
        B0.q[1] = ((const uint4*)(lb0 + kb + h * 16))[1];
        B1.q[0] = ((const uint4*)(lb1 + kb + h * 16))[0];
        B1.q[1] = ((const uint4*)(lb1 + kb + h * 16))[1];
        B2.q[0] = ((const uint4*)(lb2 + kb + h * 16))[0];
        B2.q[1] = ((const uint4*)(lb2 + kb + h * 16))[1];
        B3.q[0] = ((const uint4*)(lb3 + kb + h * 16))[0];
        B3.q[1] = ((const uint4*)(lb3 + kb + h * 16))[1];

        acc0 = __builtin_amdgcn_wmma_f32_16x16x32_bf16(false, A.v, false, B0.v, (short)0, acc0, false, false);
        acc1 = __builtin_amdgcn_wmma_f32_16x16x32_bf16(false, A.v, false, B1.v, (short)0, acc1, false, false);
        acc2 = __builtin_amdgcn_wmma_f32_16x16x32_bf16(false, A.v, false, B2.v, (short)0, acc2, false, false);
        acc3 = __builtin_amdgcn_wmma_f32_16x16x32_bf16(false, A.v, false, B3.v, (short)0, acc3, false, false);
    }

    // ---- epilogue: C layout -> VGPR r holds row (row0 + r + 8h), col = col0 + t*16 + m
#pragma unroll
    for (int t = 0; t < 4; ++t) {
        const v8f& acc = (t == 0) ? acc0 : (t == 1) ? acc1 : (t == 2) ? acc2 : acc3;
        int col = col0 + t * 16 + m;
        float b = bias[col];
#pragma unroll
        for (int r = 0; r < 8; ++r) {
            int row = row0 + r + 8 * h;
            if (row >= V) continue;
            float val = acc[r] + b;
            if (MODE == 0) {
                Yb[(size_t)row * KPAD + col] = f2bf(fmaxf(val, 0.f));
            } else {
                if (col < HIDN) Y0[(size_t)row * HIDN + col] = val;
                else            Y1[(size_t)row * HIDN + (col - HIDN)] = val;
            }
        }
    }
}

// ---------------- edge scatter: W0[dst] += W1[src] (both directions), L2 float atomics ---------
__global__ void scatter_kernel(const int* __restrict__ edges, const float* __restrict__ W1,
                               float* __restrict__ W0, int E) {
    int j = blockIdx.x;            // 0 .. 2E-1
    int s, d;
    if (j < E) { s = edges[j * 2 + 0]; d = edges[j * 2 + 1]; }
    else       { int e = j - E; s = edges[e * 2 + 1]; d = edges[e * 2 + 0]; }
    int c = threadIdx.x * 4;       // 64 threads * 4 floats = 256 channels
    const float4 w = *(const float4*)(W1 + (size_t)s * HIDN + c);
    float* dp = W0 + (size_t)d * HIDN + c;
    atomicAdd(dp + 0, w.x);
    atomicAdd(dp + 1, w.y);
    atomicAdd(dp + 2, w.z);
    atomicAdd(dp + 3, w.w);
}

// ---------------- nopos = relu(w0x + nbr) -> bf16 vfeats cols 0..255 ---------------------------
__global__ void combine_kernel(const float* __restrict__ W0, unsigned short* __restrict__ vfeats,
                               int V) {
    int idx = blockIdx.x * blockDim.x + threadIdx.x;
    if (idx >= V * HIDN) return;
    int v = idx >> 8, c = idx & 255;
    vfeats[(size_t)v * KPAD + c] = f2bf(fmaxf(W0[idx], 0.f));
}

// ---------------- output: nopos = relu(W0) in full f32 (W0 holds last w0x+nbr) -----------------
__global__ void copy_nopos(const float* __restrict__ W0, float* __restrict__ out, int V) {
    int idx = blockIdx.x * blockDim.x + threadIdx.x;
    if (idx >= V * HIDN) return;
    out[(size_t)V * 3 + idx] = fmaxf(W0[idx], 0.f);
}

// ---------------- head: new_verts = verts + tanh([relu(W0), verts] @ off_w^T + off_b) ----------
__global__ void final_kernel(const float* __restrict__ W0, const float* __restrict__ verts,
                             const float* __restrict__ off_w, const float* __restrict__ off_b,
                             float* __restrict__ out, int V) {
    int v = blockIdx.x * blockDim.x + threadIdx.x;
    if (v >= V) return;
    const float* row = W0 + (size_t)v * HIDN;
#pragma unroll
    for (int j = 0; j < 3; ++j) {
        float acc = off_b[j];
        for (int k = 0; k < HIDN; ++k) acc += off_w[j * INK + k] * fmaxf(row[k], 0.f);
#pragma unroll
        for (int k = 0; k < 3; ++k) acc += off_w[j * INK + HIDN + k] * verts[v * 3 + k];
        out[v * 3 + j] = verts[v * 3 + j] + tanhf(acc);
    }
}

static inline size_t align256(size_t x) { return (x + 255) & ~(size_t)255; }

extern "C" void kernel_launch(void* const* d_in, const int* in_sizes, int n_in,
                              void* d_out, int out_size, void* d_ws, size_t ws_size,
                              hipStream_t stream) {
    const float* x      = (const float*)d_in[0];
    const float* verts  = (const float*)d_in[1];
    const int*   edges  = (const int*)d_in[2];
    const float* bott_w = (const float*)d_in[3];
    const float* bott_b = (const float*)d_in[4];
    const float* gw0    = (const float*)d_in[5];
    const float* gb0    = (const float*)d_in[6];
    const float* gw1    = (const float*)d_in[7];
    const float* gb1    = (const float*)d_in[8];
    const float* off_w  = (const float*)d_in[9];
    const float* off_b  = (const float*)d_in[10];

    const int V = in_sizes[1] / 3;
    const int E = in_sizes[2] / 2;

    char* ws = (char*)d_ws;
    size_t o = 0;
    unsigned short* vfeats = (unsigned short*)(ws + o); o = align256(o + (size_t)V * KPAD * 2);
    float* W0 = (float*)(ws + o); o = align256(o + (size_t)V * HIDN * 4);
    float* W1 = (float*)(ws + o); o = align256(o + (size_t)V * HIDN * 4);
    unsigned short* bottbf = (unsigned short*)(ws + o); o = align256(o + (size_t)HIDN * HIDN * 2);
    unsigned short* layw   = (unsigned short*)(ws + o); o = align256(o + (size_t)3 * 512 * KPAD * 2);
    float* layb = (float*)(ws + o);
    unsigned short* sampled = (unsigned short*)W0;  // reuse: sampled dead once W0 first written
    float* outv = (float*)d_out;

    prep_weights<<<(3 * 512 * KPAD + 255) / 256, 256, 0, stream>>>(
        bott_w, gw0, gw1, gb0, gb1, bottbf, layw, layb);
    vfeats_init<<<(V + 255) / 256, 256, 0, stream>>>(verts, vfeats, V);
    sample_kernel<<<V, CCH, 0, stream>>>(x, verts, sampled, V);

    // img_feats = relu(sampled @ bott_w^T + bott_b) -> bf16 vfeats cols 0..255
    {
        dim3 g((V + 127) / 128, HIDN / 64);
        gemm_wmma<CCH, CCH, 0><<<g, 256, 0, stream>>>(sampled, bottbf, bott_b,
                                                      vfeats, nullptr, nullptr, V);
    }

    for (int l = 0; l < 3; ++l) {
        dim3 g((V + 127) / 128, 512 / 64);
        gemm_wmma<KPAD, KPAD, 1><<<g, 256, 0, stream>>>(vfeats,
                                                        layw + (size_t)l * 512 * KPAD,
                                                        layb + (size_t)l * 512,
                                                        nullptr, W0, W1, V);
        scatter_kernel<<<2 * E, 64, 0, stream>>>(edges, W1, W0, E);
        if (l < 2)
            combine_kernel<<<(V * HIDN + 255) / 256, 256, 0, stream>>>(W0, vfeats, V);
    }

    // after the last scatter, W0 holds w0x + nbr of layer 2: emit outputs in full f32
    copy_nopos<<<(V * HIDN + 255) / 256, 256, 0, stream>>>(W0, outv, V);
    final_kernel<<<(V + 127) / 128, 128, 0, stream>>>(W0, verts, off_w, off_b, outv, V);
}
